// ModernHopfieldAttention_15582141350451
// MI455X (gfx1250) — compile-verified, matrix operand order checked by reading, not run on hardware
//
#include <hip/hip_runtime.h>
#include <hip/hip_bf16.h>
#include <math.h>

// ---------------------------------------------------------------------------
// ModernHopfieldAttention for MI455X (gfx1250, wave32, WMMA).
// Compute-bound (~69 GFLOP vs ~50MB HBM) -> f16 WMMA with f32 accumulate.
// Round 2: software-pipelined GEMMs (32x64 wave tiles, double-buffered
// fragment registers) and double-buffered async-LDS flash attention so the
// WMMA pipe is never gated by s_wait_loadcnt 0 on just-issued loads.
// ---------------------------------------------------------------------------

typedef __attribute__((ext_vector_type(16))) _Float16 v16h;
typedef __attribute__((ext_vector_type(8)))  _Float16 v8h;
typedef __attribute__((ext_vector_type(8)))  float    v8f;

#define D_MODEL 1024
#define N_HEADS 16
#define DK      64
#define NB      2
#define TT      2048
#define MTOT    (NB * TT)          // 4096 rows for the projection GEMMs
#define BETA    0.125f             // 1/sqrt(64)
#define INV_BETA 8.0f

static __device__ __forceinline__ v16h join8(v8h lo, v8h hi) {
  return __builtin_shufflevector(lo, hi, 0,1,2,3,4,5,6,7,8,9,10,11,12,13,14,15);
}

static __device__ __forceinline__ v8f wmma_f16(v16h a, v16h b, v8f c) {
  // D = A(16x32 f16) x B(32x16 f16) + C(16x16 f32)
  return __builtin_amdgcn_wmma_f32_16x16x32_f16(false, a, false, b,
                                                (short)0, c, false, false);
}

// ---------------------------------------------------------------------------
// fp32 -> fp16 cast (grid-stride)
// ---------------------------------------------------------------------------
__global__ __launch_bounds__(256) void cast_to_f16(const float* __restrict__ in,
                                                   _Float16* __restrict__ out,
                                                   int n) {
  int i = blockIdx.x * blockDim.x + threadIdx.x;
  int stride = gridDim.x * blockDim.x;
  for (; i < n; i += stride) out[i] = (_Float16)in[i];
}

__global__ void zero_energy(float* e) {
  if (threadIdx.x < NB) e[threadIdx.x] = 0.0f;
}

// ---------------------------------------------------------------------------
// Core: software-pipelined 32x64 wave tile of Y = X * W^T.
// Fragments for step k+32 are loaded before the 8 WMMAs of step k issue, so
// s_wait_loadcnt covers a full k-step of latency instead of zero.
// ---------------------------------------------------------------------------
static __device__ __forceinline__ void gemm_tile_32x64(
    const _Float16* __restrict__ A, const _Float16* __restrict__ W,
    int m0, int n0, int lcol, int off8, v8f accLo[4], v8f accHi[4]) {
  const _Float16* arow0 = A + (size_t)(m0 + lcol) * D_MODEL;
  const _Float16* arow1 = arow0 + (size_t)16 * D_MODEL;
  const _Float16* wr[4];
#pragma unroll
  for (int nc = 0; nc < 4; ++nc)
    wr[nc] = W + (size_t)(n0 + nc * 16 + lcol) * D_MODEL;

  v16h a0[2], a1[2], bf[2][4];
  a0[0] = join8(*(const v8h*)(arow0 + off8), *(const v8h*)(arow0 + off8 + 16));
  a1[0] = join8(*(const v8h*)(arow1 + off8), *(const v8h*)(arow1 + off8 + 16));
#pragma unroll
  for (int nc = 0; nc < 4; ++nc)
    bf[0][nc] = join8(*(const v8h*)(wr[nc] + off8), *(const v8h*)(wr[nc] + off8 + 16));

#pragma unroll 2
  for (int kk = 0; kk < D_MODEL; kk += 32) {
    const int p = (kk >> 5) & 1, q = p ^ 1;
    if (kk + 32 < D_MODEL) {
      const int kn = kk + 32 + off8;
      a0[q] = join8(*(const v8h*)(arow0 + kn), *(const v8h*)(arow0 + kn + 16));
      a1[q] = join8(*(const v8h*)(arow1 + kn), *(const v8h*)(arow1 + kn + 16));
#pragma unroll
      for (int nc = 0; nc < 4; ++nc)
        bf[q][nc] = join8(*(const v8h*)(wr[nc] + kn), *(const v8h*)(wr[nc] + kn + 16));
    }
#pragma unroll
    for (int nc = 0; nc < 4; ++nc) {
      accLo[nc] = wmma_f16(a0[p], bf[p][nc], accLo[nc]);
      accHi[nc] = wmma_f16(a1[p], bf[p][nc], accHi[nc]);
    }
  }
}

// ---------------------------------------------------------------------------
// QKV projection: Y scattered into [b][h][t][dk] f16.
// 128 M-tiles x 16 N-tiles = 2048 wave tiles / 8 waves = 256 blocks.
// ---------------------------------------------------------------------------
__global__ __launch_bounds__(256) void qkv_gemm(const _Float16* __restrict__ A,
                                                const _Float16* __restrict__ W,
                                                const float* __restrict__ bias,
                                                _Float16* __restrict__ out) {
  const int lane = threadIdx.x & 31;
  const int wg   = blockIdx.x * 8 + (threadIdx.x >> 5);
  const int m0   = (wg >> 4) * 32;
  const int n0   = (wg & 15) * 64;
  const int lcol = lane & 15;
  const int off8 = (lane >> 4) << 3;

  v8f accLo[4] = {}, accHi[4] = {};
  gemm_tile_32x64(A, W, m0, n0, lcol, off8, accLo, accHi);

#pragma unroll
  for (int nc = 0; nc < 4; ++nc) {
    const int n = n0 + nc * 16 + lcol;
    const float bsv = bias[n];
    const int h = n >> 6, dk = n & 63;
#pragma unroll
    for (int r = 0; r < 8; ++r) {
      const int mA = m0 + r + off8;         // C-layout: M = r + 8*half
      const int mB = mA + 16;
      out[(((size_t)((mA >> 11) * N_HEADS + h) * TT) + (mA & (TT - 1))) * DK + dk] =
          (_Float16)(accLo[nc][r] + bsv);
      out[(((size_t)((mB >> 11) * N_HEADS + h) * TT) + (mB & (TT - 1))) * DK + dk] =
          (_Float16)(accHi[nc][r] + bsv);
    }
  }
}

// ---------------------------------------------------------------------------
// Output projection: f32 result straight into d_out (row-major [m][n]).
// ---------------------------------------------------------------------------
__global__ __launch_bounds__(256) void out_gemm(const _Float16* __restrict__ A,
                                                const _Float16* __restrict__ W,
                                                const float* __restrict__ bias,
                                                float* __restrict__ out) {
  const int lane = threadIdx.x & 31;
  const int wg   = blockIdx.x * 8 + (threadIdx.x >> 5);
  const int m0   = (wg >> 4) * 32;
  const int n0   = (wg & 15) * 64;
  const int lcol = lane & 15;
  const int off8 = (lane >> 4) << 3;

  v8f accLo[4] = {}, accHi[4] = {};
  gemm_tile_32x64(A, W, m0, n0, lcol, off8, accLo, accHi);

#pragma unroll
  for (int nc = 0; nc < 4; ++nc) {
    const int n = n0 + nc * 16 + lcol;
    const float bsv = bias[n];
#pragma unroll
    for (int r = 0; r < 8; ++r) {
      const int mA = m0 + r + off8;
      out[(size_t)mA * D_MODEL + n]        = accLo[nc][r] + bsv;
      out[(size_t)(mA + 16) * D_MODEL + n] = accHi[nc][r] + bsv;
    }
  }
}

// ---------------------------------------------------------------------------
// Flash attention + Hopfield energy. One block = one (b,h) x 128 query rows,
// 8 waves x 16 query rows. Key loop in tiles of 32, double-buffered:
// async K DMA + V global load for tile j+1 are issued before tile j computes.
// ---------------------------------------------------------------------------
__global__ __launch_bounds__(256) void hopfield_attn(
    const _Float16* __restrict__ Qh, const _Float16* __restrict__ Kh,
    const _Float16* __restrict__ Vh, _Float16* __restrict__ attnOut,
    float* __restrict__ energyOut) {
  __shared__ _Float16 Kt[2][32][64];      // [buf][key][dk]   8 KB
  __shared__ _Float16 Vt[2][64][32];      // [buf][dk][key]   8 KB (transposed)
  __shared__ _Float16 Pbuf[8][16][32];    // per-wave P       8 KB

  const int tid  = threadIdx.x;
  const int lane = tid & 31;
  const int w    = tid >> 5;
  const int lcol = lane & 15;
  const int off8 = (lane >> 4) << 3;

  const int qblock = blockIdx.x & 15;     // 16 q-blocks of 128 rows
  const int bh     = blockIdx.x >> 4;     // b*16 + h
  const int bidx   = bh >> 4;
  const int h      = bh & 15;

  const _Float16* Qb = Qh + (size_t)bh * TT * DK;
  const _Float16* Kb = Kh + (size_t)bh * TT * DK;
  const _Float16* Vb = Vh + (size_t)bh * TT * DK;

  // Resident Q fragments for this wave's 16 rows (dk split into 2 K-chunks)
  const int qrow = qblock * 128 + w * 16 + lcol;
  const _Float16* qp = Qb + (size_t)qrow * DK;
  const v16h aQ0 = join8(*(const v8h*)(qp + off8),      *(const v8h*)(qp + off8 + 16));
  const v16h aQ1 = join8(*(const v8h*)(qp + 32 + off8), *(const v8h*)(qp + 32 + off8 + 16));

  float mrow[8], lrow[8];
  v8f accO[4] = {};
#pragma unroll
  for (int r = 0; r < 8; ++r) { mrow[r] = -3.0e38f; lrow[r] = 0.0f; }

  // staging roles: 256 threads cover 32 keys x (64 dk / 8-per-thread)
  const int krow   = tid >> 3;            // 0..31 key row
  const int kchunk = (tid & 7) * 8;       // dk base, 16B granules
  const unsigned ldsK[2] = {(unsigned)(size_t)&Kt[0][krow][kchunk],
                            (unsigned)(size_t)&Kt[1][krow][kchunk]};
  const unsigned long long kbase = (unsigned long long)Kb;

  // ---- prologue: stage tile 0 into buffer 0 ----
  {
    unsigned gOff = (unsigned)((krow * DK + kchunk) * 2);
    asm volatile("global_load_async_to_lds_b128 %0, %1, %2"
                 :: "v"(ldsK[0]), "v"(gOff), "s"(kbase) : "memory");
    v8h vv = *(const v8h*)(Vb + (size_t)krow * DK + kchunk);
#pragma unroll
    for (int e = 0; e < 8; ++e) Vt[0][kchunk + e][krow] = vv[e];
    asm volatile("s_wait_asynccnt 0" ::: "memory");
  }
  __syncthreads();

  for (int jt = 0; jt < TT / 32; ++jt) {
    const int cur = jt & 1, nxt = cur ^ 1;
    const bool more = (jt + 1) < (TT / 32);

    // --- issue next tile's loads before computing this tile ---
    v8h vvn;
    if (more) {
      const int jn = (jt + 1) * 32;
      unsigned gOff = (unsigned)(((jn + krow) * DK + kchunk) * 2);
      asm volatile("global_load_async_to_lds_b128 %0, %1, %2"
                   :: "v"(ldsK[nxt]), "v"(gOff), "s"(kbase) : "memory");
      vvn = *(const v8h*)(Vb + (size_t)(jn + krow) * DK + kchunk);
    }

    // --- S = beta * Q K^T for 32 keys (two 16x16 C-tiles, 4 WMMAs) ---
    v8f s0 = {}, s1 = {};
    {
      const _Float16* k0 = &Kt[cur][lcol][0];
      const _Float16* k1 = &Kt[cur][16 + lcol][0];
      v16h b00 = join8(*(const v8h*)(k0 + off8),      *(const v8h*)(k0 + off8 + 16));
      v16h b01 = join8(*(const v8h*)(k0 + 32 + off8), *(const v8h*)(k0 + 32 + off8 + 16));
      v16h b10 = join8(*(const v8h*)(k1 + off8),      *(const v8h*)(k1 + off8 + 16));
      v16h b11 = join8(*(const v8h*)(k1 + 32 + off8), *(const v8h*)(k1 + 32 + off8 + 16));
      s0 = wmma_f16(aQ0, b00, s0);
      s0 = wmma_f16(aQ1, b01, s0);
      s1 = wmma_f16(aQ0, b10, s1);
      s1 = wmma_f16(aQ1, b11, s1);
    }

    // --- online softmax (row stats replicated across each 16-lane half) ---
    float corr[8];
#pragma unroll
    for (int r = 0; r < 8; ++r) {
      float x0 = s0[r] * BETA;
      float x1 = s1[r] * BETA;
      float mx = fmaxf(x0, x1);
#pragma unroll
      for (int d = 1; d < 16; d <<= 1) mx = fmaxf(mx, __shfl_xor(mx, d, 16));
      float mnew = fmaxf(mrow[r], mx);
      corr[r] = __expf(mrow[r] - mnew);
      float p0 = __expf(x0 - mnew);
      float p1 = __expf(x1 - mnew);
      float ps = p0 + p1;
#pragma unroll
      for (int d = 1; d < 16; d <<= 1) ps += __shfl_xor(ps, d, 16);
      lrow[r] = lrow[r] * corr[r] + ps;
      mrow[r] = mnew;
      // spill P in C-layout so it can be re-read in A-fragment layout
      Pbuf[w][r + off8][lcol]      = (_Float16)p0;
      Pbuf[w][r + off8][16 + lcol] = (_Float16)p1;
    }
#pragma unroll
    for (int c = 0; c < 4; ++c)
#pragma unroll
      for (int r = 0; r < 8; ++r) accO[c][r] *= corr[r];

    asm volatile("s_wait_dscnt 0" ::: "memory");

    // --- O += P(16x32) x V(32x64): 4 WMMAs over dk chunks of 16 ---
    {
      const _Float16* pp = &Pbuf[w][lcol][0];
      v16h aP = join8(*(const v8h*)(pp + off8), *(const v8h*)(pp + off8 + 16));
#pragma unroll
      for (int c = 0; c < 4; ++c) {
        const _Float16* vp = &Vt[cur][c * 16 + lcol][0];
        v16h bV = join8(*(const v8h*)(vp + off8), *(const v8h*)(vp + off8 + 16));
        accO[c] = wmma_f16(aP, bV, accO[c]);
      }
    }

    // --- finish next tile's staging (V scatter after compute hid the load) ---
    if (more) {
#pragma unroll
      for (int e = 0; e < 8; ++e) Vt[nxt][kchunk + e][krow] = vvn[e];
      asm volatile("s_wait_asynccnt 0" ::: "memory");
    }
    __syncthreads();
  }

  // --- epilogue: normalize, write attn output (f16 [b][t][h*64+dk]) ---
  float inv[8];
#pragma unroll
  for (int r = 0; r < 8; ++r) inv[r] = 1.0f / lrow[r];
#pragma unroll
  for (int c = 0; c < 4; ++c) {
#pragma unroll
    for (int r = 0; r < 8; ++r) {
      const int t = qblock * 128 + w * 16 + r + off8;
      attnOut[((size_t)(bidx * TT + t)) * D_MODEL + h * DK + c * 16 + lcol] =
          (_Float16)(accO[c][r] * inv[r]);
    }
  }

  // --- Hopfield energy: -lse/beta summed; lanes 0 and 16 own rows 0-7 / 8-15
  if (lcol == 0) {
    float e = 0.0f;
#pragma unroll
    for (int r = 0; r < 8; ++r) e -= (mrow[r] + __logf(lrow[r])) * INV_BETA;
    atomicAdd(&energyOut[bidx], e);
  }
}

// ---------------------------------------------------------------------------
extern "C" void kernel_launch(void* const* d_in, const int* in_sizes, int n_in,
                              void* d_out, int out_size, void* d_ws, size_t ws_size,
                              hipStream_t stream) {
  const float* x  = (const float*)d_in[0];
  const float* Wq = (const float*)d_in[1];
  const float* bq = (const float*)d_in[2];
  const float* Wk = (const float*)d_in[3];
  const float* bk = (const float*)d_in[4];
  const float* Wv = (const float*)d_in[5];
  const float* bv = (const float*)d_in[6];
  const float* Wo = (const float*)d_in[7];
  const float* bo = (const float*)d_in[8];

  float* out    = (float*)d_out;
  float* energy = out + (size_t)NB * TT * D_MODEL;   // [B] after out tensor

  const size_t MB = 1u << 20;
  if (ws_size < 48 * MB) return;   // need 48 MiB of scratch
  char* ws = (char*)d_ws;
  _Float16* xh  = (_Float16*)(ws);             // 8 MiB
  _Float16* wqh = (_Float16*)(ws +  8 * MB);   // 2 MiB each
  _Float16* wkh = (_Float16*)(ws + 10 * MB);
  _Float16* wvh = (_Float16*)(ws + 12 * MB);
  _Float16* woh = (_Float16*)(ws + 14 * MB);
  _Float16* Qh  = (_Float16*)(ws + 16 * MB);   // 8 MiB each, [b][h][t][dk]
  _Float16* Kh  = (_Float16*)(ws + 24 * MB);
  _Float16* Vh  = (_Float16*)(ws + 32 * MB);
  _Float16* Ah  = (_Float16*)(ws + 40 * MB);   // attn out, [b][t][h*dk]

  cast_to_f16<<<1024, 256, 0, stream>>>(x,  xh,  MTOT * D_MODEL);
  cast_to_f16<<<512,  256, 0, stream>>>(Wq, wqh, D_MODEL * D_MODEL);
  cast_to_f16<<<512,  256, 0, stream>>>(Wk, wkh, D_MODEL * D_MODEL);
  cast_to_f16<<<512,  256, 0, stream>>>(Wv, wvh, D_MODEL * D_MODEL);
  cast_to_f16<<<512,  256, 0, stream>>>(Wo, woh, D_MODEL * D_MODEL);
  zero_energy<<<1, 32, 0, stream>>>(energy);

  qkv_gemm<<<256, 256, 0, stream>>>(xh, wqh, bq, Qh);
  qkv_gemm<<<256, 256, 0, stream>>>(xh, wkh, bk, Kh);
  qkv_gemm<<<256, 256, 0, stream>>>(xh, wvh, bv, Vh);

  hopfield_attn<<<NB * N_HEADS * (TT / 128), 256, 0, stream>>>(Qh, Kh, Vh, Ah, energy);

  out_gemm<<<256, 256, 0, stream>>>(Ah, woh, bo, out);
}
